// ParticleAttentionBlock_2551210574403
// MI455X (gfx1250) — compile-verified, matrix operand order checked by reading, not run on hardware
//
#include <hip/hip_runtime.h>
#include <hip/hip_bf16.h>

typedef __bf16 bf16;
typedef __attribute__((ext_vector_type(16))) __bf16 v16bf;
typedef __attribute__((ext_vector_type(8)))  __bf16 v8bf;
typedef __attribute__((ext_vector_type(8)))  float  v8f;

#define BB 4
#define NT 256
#define CC 512
#define HH 8
#define DD 64

union ABFrag { v16bf v; v8bf h[2]; };

// Load a 16-element bf16 A/B fragment slice for lane: two 16-byte chunks at
// koff and koff+16 (ISA 7.12.2 16-bit operand layout).
__device__ __forceinline__ v16bf load_frag(const bf16* __restrict__ rowbase, int koff) {
    ABFrag f;
    f.h[0] = *reinterpret_cast<const v8bf*>(rowbase + koff);
    f.h[1] = *reinterpret_cast<const v8bf*>(rowbase + koff + 16);
    return f.v;
}
__device__ __forceinline__ v16bf load_frag_lds(const bf16* rowbase, int koff) {
    ABFrag f;
    f.h[0] = *reinterpret_cast<const v8bf*>(rowbase + koff);
    f.h[1] = *reinterpret_cast<const v8bf*>(rowbase + koff + 16);
    return f.v;
}

__device__ __forceinline__ v8f wmma_bf16(v16bf a, v16bf b, v8f c) {
    return __builtin_amdgcn_wmma_f32_16x16x32_bf16(false, a, false, b, (short)0, c, false, false);
}

// ---- gfx1250 async global -> LDS (VGLOBAL opcode 98, tracked by ASYNCcnt) ----
__device__ __forceinline__ void async_load_b128(unsigned int lds_off, const void* gptr) {
    asm volatile("global_load_async_to_lds_b128 %0, %1, off"
                 :: "v"(lds_off), "v"(gptr) : "memory");
}
#define WAIT_ASYNCCNT(n) asm volatile("s_wait_asynccnt " #n ::: "memory")

// ---------------- fp32 -> bf16 convert ----------------
__global__ __launch_bounds__(256) void cvt_bf16_kernel(const float* __restrict__ in,
                                                       bf16* __restrict__ out, int n) {
    int t = blockIdx.x * 256 + threadIdx.x;
    if (t < n) out[t] = (bf16)in[t];
}

// ---------------- LayerNorm -> bf16 ----------------
__global__ __launch_bounds__(256) void ln_kernel(const float* __restrict__ x,
                                                 const float* __restrict__ g,
                                                 const float* __restrict__ b,
                                                 bf16* __restrict__ out) {
    __shared__ float rs[256], rs2[256];
    int row = blockIdx.x;
    const float* xr = x + (size_t)row * CC;
    float s = 0.f, s2 = 0.f;
    for (int c = threadIdx.x; c < CC; c += 256) { float v = xr[c]; s += v; s2 += v * v; }
    rs[threadIdx.x] = s; rs2[threadIdx.x] = s2;
    __syncthreads();
    for (int st = 128; st > 0; st >>= 1) {
        if (threadIdx.x < st) { rs[threadIdx.x] += rs[threadIdx.x + st]; rs2[threadIdx.x] += rs2[threadIdx.x + st]; }
        __syncthreads();
    }
    float mean = rs[0] * (1.0f / CC);
    float var  = rs2[0] * (1.0f / CC) - mean * mean;
    float rstd = rsqrtf(var + 1e-5f);
    for (int c = threadIdx.x; c < CC; c += 256)
        out[(size_t)row * CC + c] = (bf16)((xr[c] - mean) * rstd * g[c] + b[c]);
}

// ---------------- generic bf16 WMMA GEMM: out = act(X @ W^T + bias (+resid)) ----
// X: [M,K] bf16, W: [Nout,K] bf16. Wave computes a 32x64 tile: 2 A fragments,
// 8 accumulators, each B fragment reused across both A rows (halves B traffic).
// Block = 128 threads = 4 waves -> block tile 32 x 256.
__global__ __launch_bounds__(128) void gemm_bf16_kernel(
    const bf16* __restrict__ X, const bf16* __restrict__ W,
    const float* __restrict__ bias, const float* __restrict__ resid,
    bf16* __restrict__ out_bf, float* __restrict__ out_f32,
    int M, int K, int Nout, int act) {
    const int lane = threadIdx.x & 31;
    const int wave = threadIdx.x >> 5;
    const int m0 = blockIdx.x * 32;
    const int n0 = blockIdx.y * 256 + wave * 64;
    const int lrow  = lane & 15;
    const int koff0 = (lane >> 4) * 8;

    v8f acc[2][4] = {};
    const bf16* xrow0 = X + (size_t)(m0 + lrow) * K;
    const bf16* xrow1 = X + (size_t)(m0 + 16 + lrow) * K;
    const bf16* wrow0 = W + (size_t)(n0 + lrow) * K;

    for (int k0 = 0; k0 < K; k0 += 32) {
        __builtin_prefetch(xrow0 + k0 + 128, 0, 3);
        v16bf a0 = load_frag(xrow0, k0 + koff0);
        v16bf a1 = load_frag(xrow1, k0 + koff0);
#pragma unroll
        for (int j = 0; j < 4; ++j) {
            v16bf bm = load_frag(wrow0 + (size_t)(j * 16) * K, k0 + koff0);
            acc[0][j] = wmma_bf16(a0, bm, acc[0][j]);
            acc[1][j] = wmma_bf16(a1, bm, acc[1][j]);
        }
    }
#pragma unroll
    for (int gi = 0; gi < 2; ++gi) {
        const int rbase = m0 + gi * 16 + ((lane >> 4) << 3);
#pragma unroll
        for (int j = 0; j < 4; ++j) {
            int col = n0 + j * 16 + lrow;
            float bv = bias ? bias[col] : 0.f;
#pragma unroll
            for (int e = 0; e < 8; ++e) {
                int r = rbase + e;
                float v = acc[gi][j][e] + bv;
                if (resid) v += resid[(size_t)r * Nout + col];
                if (act == 1) v = 0.5f * v * (1.0f + erff(v * 0.70710678118f));  // exact GELU
                if (out_f32) out_f32[(size_t)r * Nout + col] = v;
                else         out_bf [(size_t)r * Nout + col] = (bf16)v;
            }
        }
    }
}

// ---------------- pair bias: bias[b,h,i,j] = u[b,i,j,:]·pair_w[h,:] + pair_b[h] ---
// Memory-bound: streams 512 MB of u_ij through async global->LDS double
// buffering (ASYNCcnt-pipelined; in-flight data costs no VGPRs).
// Block = 256 threads = 8 waves; batch = 8 rows (16 KB); PB_BATCHES batches/block.
#define PB_BATCHES 16
__global__ __launch_bounds__(256) void pair_bias_kernel(
    const float* __restrict__ u, const float* __restrict__ pw,
    const float* __restrict__ pb, const unsigned char* __restrict__ mask,
    float* __restrict__ bias) {
    __shared__ float lpw[HH * CC];                       // 16 KB
    __shared__ __align__(16) float ubuf[2][8][CC];      // 2 x 16 KB ping-pong
    for (int t = threadIdx.x; t < HH * CC; t += 256) lpw[t] = pw[t];

    const int tid  = threadIdx.x;
    const int lane = threadIdx.x & 31;
    const int wave = threadIdx.x >> 5;
    const long blockRow0 = (long)blockIdx.x * (PB_BATCHES * 8);   // over B*N*N

    // issue batch 0: 256 threads x 4 x 16B = 16 KB
    {
        const char* g = (const char*)(u + blockRow0 * CC) + tid * 64;
        unsigned int l = (unsigned int)(size_t)((char*)&ubuf[0][0][0] + tid * 64);
#pragma unroll
        for (int q = 0; q < 4; ++q) async_load_b128(l + q * 16, g + q * 16);
    }

    for (int it = 0; it < PB_BATCHES; ++it) {
        if (it + 1 < PB_BATCHES) {
            const char* g = (const char*)(u + (blockRow0 + (long)(it + 1) * 8) * CC) + tid * 64;
            unsigned int l = (unsigned int)(size_t)((char*)&ubuf[(it + 1) & 1][0][0] + tid * 64);
#pragma unroll
            for (int q = 0; q < 4; ++q) async_load_b128(l + q * 16, g + q * 16);
            WAIT_ASYNCCNT(0x4);     // async loads complete in order: batch `it` landed
        } else {
            WAIT_ASYNCCNT(0x0);
        }
        __syncthreads();            // batch `it` visible to whole block (incl. lpw on it==0)

        // compute: one row per wave, 16 c's per lane
        long rowIdx = blockRow0 + (long)it * 8 + wave;
        int j = (int)(rowIdx & 255);
        int i = (int)((rowIdx >> 8) & 255);
        int b = (int)(rowIdx >> 16);
        const float4* row4 = reinterpret_cast<const float4*>(&ubuf[it & 1][wave][0]) + lane * 4;
        float acc[HH] = {};
#pragma unroll
        for (int q = 0; q < 4; ++q) {
            float4 uv = row4[q];
            int c = lane * 16 + q * 4;
#pragma unroll
            for (int h = 0; h < HH; ++h) {
                const float* w = lpw + h * CC + c;
                acc[h] += uv.x * w[0] + uv.y * w[1] + uv.z * w[2] + uv.w * w[3];
            }
        }
#pragma unroll
        for (int h = 0; h < HH; ++h) {
            float v = acc[h];
            for (int off = 16; off > 0; off >>= 1) v += __shfl_xor(v, off, 32);
            if (lane == h) {
                v += pb[h];
                if (!mask[b * NT + j]) v = -__builtin_inff();
                bias[(((size_t)b * HH + h) * NT + i) * NT + j] = v;
            }
        }
        __syncthreads();            // done reading ubuf[it&1] before batch it+2 overwrites it
    }
}

// ---------------- repack V -> V^T : vt[bh][d][m] ----------------
__global__ __launch_bounds__(256) void build_vt_kernel(const bf16* __restrict__ qkv,
                                                       bf16* __restrict__ vt) {
    int t = blockIdx.x * 256 + threadIdx.x;            // B*H*D*N
    int m = t & 255;
    int d = (t >> 8) & 63;
    int h = (t >> 14) & 7;
    int b = t >> 17;
    vt[t] = qkv[(size_t)(b * NT + m) * (3 * CC) + 2 * CC + h * DD + d];
}

// ---------------- attention: per (b,h,16-query tile) --------------------------
// S = (Q K^T)/8 + bias -> softmax -> out = P V, all via WMMA bf16.
__global__ __launch_bounds__(128) void attn_kernel(
    const bf16* __restrict__ qkv,     // [B*N][3C]
    const bf16* __restrict__ vt,      // [B*H][D][N]
    const float* __restrict__ bias,   // [B*H][N][N]
    bf16* __restrict__ outbf) {       // [B*N][C]
    __shared__ __align__(16) float S[16][NT];
    __shared__ __align__(16) bf16  P[16][NT];
    __shared__ float red[16][8];

    const int bh = blockIdx.y;        // 0..31
    const int b  = bh >> 3, h = bh & 7;
    const int qt = blockIdx.x;        // 0..15
    const int lane = threadIdx.x & 31;
    const int wave = threadIdx.x >> 5;
    const int lrow  = lane & 15;
    const int koff0 = (lane >> 4) * 8;

    // Q fragments (K = D = 64 -> two K=32 steps)
    const bf16* qbase = qkv + (size_t)(b * NT + qt * 16 + lrow) * (3 * CC) + h * DD;
    v16bf aq0 = load_frag(qbase, 0 + koff0);
    v16bf aq1 = load_frag(qbase, 32 + koff0);
    const float* biasq = bias + ((size_t)bh * NT + qt * 16) * NT;

    // scores: 16 column tiles, 4 waves round-robin
    for (int ct = wave; ct < 16; ct += 4) {
        const bf16* kbase = qkv + (size_t)(b * NT + ct * 16 + lrow) * (3 * CC) + CC + h * DD;
        v16bf bk0 = load_frag(kbase, 0 + koff0);
        v16bf bk1 = load_frag(kbase, 32 + koff0);
        v8f acc = {};
        acc = wmma_bf16(aq0, bk0, acc);
        acc = wmma_bf16(aq1, bk1, acc);
        const int rb = (lane >> 4) << 3;
        const int col = ct * 16 + lrow;
#pragma unroll
        for (int e = 0; e < 8; ++e) {
            int r = rb + e;
            S[r][col] = acc[e] * 0.125f + biasq[(size_t)r * NT + col];
        }
    }
    __syncthreads();

    // softmax: 8 threads per row, 32 columns each
    const int rown = threadIdx.x >> 3;
    const int sub  = threadIdx.x & 7;
    float mx = -__builtin_inff();
    for (int c = sub * 32; c < sub * 32 + 32; ++c) mx = fmaxf(mx, S[rown][c]);
    red[rown][sub] = mx;
    __syncthreads();
    if (sub == 0) {
        float m = red[rown][0];
        for (int i = 1; i < 8; ++i) m = fmaxf(m, red[rown][i]);
        red[rown][0] = m;
    }
    __syncthreads();
    mx = red[rown][0];
    float sum = 0.f;
    for (int c = sub * 32; c < sub * 32 + 32; ++c) {
        float e = __expf(S[rown][c] - mx);
        S[rown][c] = e;
        sum += e;
    }
    __syncthreads();
    red[rown][sub] = sum;
    __syncthreads();
    if (sub == 0) {
        float s = 0.f;
        for (int i = 0; i < 8; ++i) s += red[rown][i];
        red[rown][0] = 1.0f / s;
    }
    __syncthreads();
    float inv = red[rown][0];
    for (int c = sub * 32; c < sub * 32 + 32; ++c) P[rown][c] = (bf16)(S[rown][c] * inv);
    __syncthreads();

    // out = P @ V : wave handles a 16-wide d-tile; K = N = 256 -> 8 steps
    v8f acc = {};
    const bf16* vtb = vt + ((size_t)bh * DD + wave * 16 + lrow) * NT;
    for (int k0 = 0; k0 < NT; k0 += 32) {
        v16bf a  = load_frag_lds(&P[lrow][0], k0 + koff0);
        v16bf bv = load_frag(vtb, k0 + koff0);
        acc = wmma_bf16(a, bv, acc);
    }
    const int rb = (lane >> 4) << 3;
#pragma unroll
    for (int e = 0; e < 8; ++e) {
        int tok = qt * 16 + rb + e;
        int col = h * DD + wave * 16 + lrow;
        outbf[(size_t)(b * NT + tok) * CC + col] = (bf16)acc[e];
    }
}

// =============================================================================
extern "C" void kernel_launch(void* const* d_in, const int* in_sizes, int n_in,
                              void* d_out, int out_size, void* d_ws, size_t ws_size,
                              hipStream_t stream) {
    const float* x      = (const float*)d_in[0];
    const float* u_ij   = (const float*)d_in[1];
    const unsigned char* pmask = (const unsigned char*)d_in[2];
    const float* qkv_w  = (const float*)d_in[3];
    const float* proj_w = (const float*)d_in[4];
    const float* proj_b = (const float*)d_in[5];
    const float* ln1_g  = (const float*)d_in[6];
    const float* ln1_b  = (const float*)d_in[7];
    const float* ln2_g  = (const float*)d_in[8];
    const float* ln2_b  = (const float*)d_in[9];
    const float* w1     = (const float*)d_in[10];
    const float* b1     = (const float*)d_in[11];
    const float* w2     = (const float*)d_in[12];
    const float* b2     = (const float*)d_in[13];
    const float* pair_w = (const float*)d_in[14];
    const float* pair_b = (const float*)d_in[15];
    float* out = (float*)d_out;

    const int M = BB * NT;  // 1024
    char* ws = (char*)d_ws;
    size_t off = 0;
    auto alloc = [&](size_t bytes) { void* p = ws + off; off = (off + bytes + 255) & ~(size_t)255; return p; };

    bf16* qkv_w_bf = (bf16*)alloc((size_t)3 * CC * CC * 2);
    bf16* proj_w_bf = (bf16*)alloc((size_t)CC * CC * 2);
    bf16* w1_bf = (bf16*)alloc((size_t)4 * CC * CC * 2);
    bf16* w2_bf = (bf16*)alloc((size_t)4 * CC * CC * 2);
    bf16* xn_bf = (bf16*)alloc((size_t)M * CC * 2);
    bf16* qkv_bf = (bf16*)alloc((size_t)M * 3 * CC * 2);
    float* bias_f = (float*)alloc((size_t)BB * HH * NT * NT * 4);
    bf16* vt = (bf16*)alloc((size_t)BB * HH * DD * NT * 2);
    bf16* ao_bf = (bf16*)alloc((size_t)M * CC * 2);
    float* x1 = (float*)alloc((size_t)M * CC * 4);
    bf16* xn2_bf = (bf16*)alloc((size_t)M * CC * 2);
    bf16* h_bf = (bf16*)alloc((size_t)M * 4 * CC * 2);

    // 1) weights -> bf16
    cvt_bf16_kernel<<<(3 * CC * CC + 255) / 256, 256, 0, stream>>>(qkv_w, qkv_w_bf, 3 * CC * CC);
    cvt_bf16_kernel<<<(CC * CC + 255) / 256, 256, 0, stream>>>(proj_w, proj_w_bf, CC * CC);
    cvt_bf16_kernel<<<(4 * CC * CC + 255) / 256, 256, 0, stream>>>(w1, w1_bf, 4 * CC * CC);
    cvt_bf16_kernel<<<(4 * CC * CC + 255) / 256, 256, 0, stream>>>(w2, w2_bf, 4 * CC * CC);

    // 2) LN1
    ln_kernel<<<M, 256, 0, stream>>>(x, ln1_g, ln1_b, xn_bf);

    // 3) QKV = xn @ qkv_w^T   (M=1024, K=512, N=1536)
    gemm_bf16_kernel<<<dim3(M / 32, (3 * CC) / 256), 128, 0, stream>>>(
        xn_bf, qkv_w_bf, nullptr, nullptr, qkv_bf, nullptr, M, CC, 3 * CC, 0);

    // 4) pair bias (streams the 512 MB u_ij via async global->LDS)
    pair_bias_kernel<<<(BB * NT * NT) / (PB_BATCHES * 8), 256, 0, stream>>>(
        u_ij, pair_w, pair_b, pmask, bias_f);

    // 5) V^T repack
    build_vt_kernel<<<(BB * HH * DD * NT) / 256, 256, 0, stream>>>(qkv_bf, vt);

    // 6) attention
    attn_kernel<<<dim3(NT / 16, BB * HH), 128, 0, stream>>>(qkv_bf, vt, bias_f, ao_bf);

    // 7) x1 = attn_out @ proj_w^T + proj_b + x
    gemm_bf16_kernel<<<dim3(M / 32, CC / 256), 128, 0, stream>>>(
        ao_bf, proj_w_bf, proj_b, x, nullptr, x1, M, CC, CC, 0);

    // 8) LN2
    ln_kernel<<<M, 256, 0, stream>>>(x1, ln2_g, ln2_b, xn2_bf);

    // 9) h = gelu(xn2 @ w1^T + b1)   (N=2048)
    gemm_bf16_kernel<<<dim3(M / 32, (4 * CC) / 256), 128, 0, stream>>>(
        xn2_bf, w1_bf, b1, nullptr, h_bf, nullptr, M, CC, 4 * CC, 1);

    // 10) out = h @ w2^T + b2 + x1   (K=2048)
    gemm_bf16_kernel<<<dim3(M / 32, CC / 256), 128, 0, stream>>>(
        h_bf, w2_bf, b2, x1, nullptr, out, M, 4 * CC, CC, 0);
}
